// LearnedSparseAttention_35974646071447
// MI455X (gfx1250) — matrix operand order repaired and checked by reading, not compiled, not run on hardware
//
#include <hip/hip_runtime.h>

typedef __attribute__((ext_vector_type(16))) __bf16 v16bf;
typedef __attribute__((ext_vector_type(8)))  __bf16 v8bf;
typedef __attribute__((ext_vector_type(8)))  float  v8f;
typedef __attribute__((ext_vector_type(4)))  unsigned int u32x4;
typedef __attribute__((ext_vector_type(8)))  int i32x8;
typedef __attribute__((ext_vector_type(4)))  int i32x4;

#define NEGINF (-__builtin_inff())

static __device__ __forceinline__ v8f wmma_bf16(v16bf a, v16bf b, v8f c) {
  return __builtin_amdgcn_wmma_f32_16x16x32_bf16(false, a, false, b, (short)0, c,
                                                 false, false);
}

static __device__ __forceinline__ v16bf ld_row16(const __bf16* p) {
  v8bf lo = *reinterpret_cast<const v8bf*>(p);
  v8bf hi = *reinterpret_cast<const v8bf*>(p + 8);
  return __builtin_shufflevector(lo, hi, 0, 1, 2, 3, 4, 5, 6, 7,
                                 8, 9, 10, 11, 12, 13, 14, 15);
}

static __device__ __forceinline__ v16bf ld_afrag(const __bf16* row_ptr) {
  v8bf lo = *reinterpret_cast<const v8bf*>(row_ptr);
  v8bf hi = *reinterpret_cast<const v8bf*>(row_ptr + 8);
  return __builtin_shufflevector(lo, hi, 0, 1, 2, 3, 4, 5, 6, 7,
                                 8, 9, 10, 11, 12, 13, 14, 15);
}

static __device__ __forceinline__ unsigned lds_off(const void* p) {
  return (unsigned)(size_t)p;  // generic LDS addr: low 32 bits = LDS offset
}

// Tensor Data Mover: load a (tile_d1 x tile_d0) tile (row stride in elements)
// of 2-byte elements from global into LDS. tile_d1==0 -> 1D copy.
// D# packing per CDNA5 ISA 8.3/8.4 (count=1, type=2=image, data_size=1 (2B)).
static __device__ __forceinline__ void tdm_load_to_lds(
    unsigned ldsaddr, const void* gsrc, unsigned tile_d0, unsigned tile_d1,
    unsigned long long row_stride, unsigned tensor_d0, unsigned tensor_d1) {
  unsigned long long ga = (unsigned long long)(size_t)gsrc;
  u32x4 g0;
  g0[0] = 1u;                                    // count=1, user descriptor
  g0[1] = ldsaddr;                               // lds_addr [63:32]
  g0[2] = (unsigned)ga;                          // global_addr lo
  g0[3] = (unsigned)((ga >> 32) & 0x1FFFFFFull)  // global_addr [56:32]
          | (2u << 30);                          // type=2
  i32x8 g1;
  g1[0] = (int)(1u << 16);                                        // data_size=2B
  g1[1] = (int)((tensor_d0 & 0xFFFFu) << 16);                     // td0[15:0]
  g1[2] = (int)((tensor_d0 >> 16) | ((tensor_d1 & 0xFFFFu) << 16));
  g1[3] = (int)((tensor_d1 >> 16) | ((tile_d0 & 0xFFFFu) << 16)); // tile_dim0
  g1[4] = (int)(tile_d1 & 0xFFFFu);                               // tile_dim1
  g1[5] = (int)(row_stride & 0xFFFFFFFFull);                      // dim0 stride lo
  g1[6] = (int)((row_stride >> 32) & 0xFFFFull);                  // dim0 stride hi
  g1[7] = 0;                                                      // dim1 stride = 0
  i32x4 z4 = {0, 0, 0, 0};
#if __clang_major__ >= 23
  i32x8 z8 = {0, 0, 0, 0, 0, 0, 0, 0};
  __builtin_amdgcn_tensor_load_to_lds(g0, g1, z4, z4, z8, 0);
#else
  __builtin_amdgcn_tensor_load_to_lds(g0, g1, z4, z4, 0);
#endif
}

__global__ void cvt_f32_bf16_kernel(const float* __restrict__ in,
                                    __bf16* __restrict__ out, int n) {
  int i = blockIdx.x * blockDim.x + threadIdx.x;
  if (i < n) out[i] = (__bf16)in[i];
}

// C[m][n] = sum_k A[m][k]*W[n][k]; 32x64 tile/wave, register double-buffered.
// Epilogue scatters q/k (B,H,T,dh) bf16 and v transposed (B,H,dh,T) bf16.
__global__ __launch_bounds__(256) void gemm_qkv_kernel(
    const __bf16* __restrict__ A, const __bf16* __restrict__ W,
    __bf16* __restrict__ qw, __bf16* __restrict__ kw, __bf16* __restrict__ vtw) {
  const int K = 1024, NT = 48, T = 2048, H = 16, DH = 64;
  int w = blockIdx.x * 8 + (threadIdx.x >> 5);
  int mt = w / NT, nt = w % NT;
  int lane = threadIdx.x & 31;
  int half = lane >> 4, l15 = lane & 15;
  int m0 = mt * 32, n0 = nt * 64;

  const __bf16* a0p = A + (size_t)(m0 + l15) * K + half * 16;
  const __bf16* a1p = a0p + (size_t)16 * K;
  const __bf16* bp[4];
#pragma unroll
  for (int t4 = 0; t4 < 4; ++t4)
    bp[t4] = W + (size_t)(n0 + t4 * 16 + l15) * K + half * 16;

  v8f acc[8] = {};
  v16bf a0c = ld_afrag(a0p), a1c = ld_afrag(a1p);
  v16bf bc[4];
#pragma unroll
  for (int t4 = 0; t4 < 4; ++t4) bc[t4] = ld_row16(bp[t4]);

  for (int k0 = 0; k0 < K; k0 += 32) {
    int kn = (k0 + 32 < K) ? k0 + 32 : 0;  // last-iter reload is harmless
    v16bf a0n = ld_afrag(a0p + kn);
    v16bf a1n = ld_afrag(a1p + kn);
    v16bf bn[4];
#pragma unroll
    for (int t4 = 0; t4 < 4; ++t4) bn[t4] = ld_row16(bp[t4] + kn);
#pragma unroll
    for (int t4 = 0; t4 < 4; ++t4) {
      acc[t4] = wmma_bf16(a0c, bc[t4], acc[t4]);
      acc[4 + t4] = wmma_bf16(a1c, bc[t4], acc[4 + t4]);
    }
    a0c = a0n; a1c = a1n;
#pragma unroll
    for (int t4 = 0; t4 < 4; ++t4) bc[t4] = bn[t4];
  }

  int b = m0 >> 11;
  int sect = n0 >> 10;
  int h = (n0 & 1023) >> 6;
  int bh = b * H + h;
#pragma unroll
  for (int mi = 0; mi < 2; ++mi) {
#pragma unroll
    for (int t4 = 0; t4 < 4; ++t4) {
#pragma unroll
      for (int r = 0; r < 8; ++r) {
        int row = m0 + mi * 16 + r + half * 8;
        int t = row & (T - 1);
        int dh = t4 * 16 + l15;
        __bf16 val = (__bf16)acc[mi * 4 + t4][r];
        if (sect == 0)      qw[(bh * T + t) * DH + dh] = val;
        else if (sect == 1) kw[(bh * T + t) * DH + dh] = val;
        else                vtw[(bh * DH + dh) * T + t] = val;
      }
    }
  }
}

// One wave per 16-query tile. TDM double-buffers 32-key K chunks (1D) and
// 64x32 V chunks (2D) into LDS, overlapped with WMMA via TENSORcnt.
__global__ __launch_bounds__(32) void attn_topk_kernel(
    const __bf16* __restrict__ qw, const __bf16* __restrict__ kw,
    const __bf16* __restrict__ vtw, __bf16* __restrict__ ao) {
  const int T = 2048, H = 16, DH = 64, D = 1024;
  const int nQT = T / 16;
  int tile = blockIdx.x;
  int qt = tile % nQT;
  int bh = tile / nQT;
  int b = bh / H, h = bh % H;
  int lane = threadIdx.x;
  int half = lane >> 4, l15 = lane & 15;
  int m0 = qt * 16;

  const __bf16* qb = qw + (size_t)bh * T * DH;
  const __bf16* kb = kw + (size_t)bh * T * DH;
  const __bf16* vb = vtw + (size_t)bh * DH * T;

  __shared__ __align__(16) __bf16 kt[2][32 * 64];   // 2 x 4KB key chunks
  __shared__ __align__(16) __bf16 vt[2][64 * 32];   // 2 x 4KB V chunks
  __shared__ __align__(16) float sm_s[16][33];
  __shared__ float sm_thr[16];
  __shared__ float sm_max[16];
  __shared__ __align__(16) __bf16 sp[16][32];

  const __bf16* qrow = qb + (size_t)(m0 + l15) * DH + half * 16;
  v16bf qa0 = ld_afrag(qrow);
  v16bf qa1 = ld_afrag(qrow + 32);

  int npair = (qt >> 1) + 1;  // 32-key chunks; tail masked causally

  // ---------- Pass A: per-row top-8 of causal scaled scores ----------
  float top[8];
#pragma unroll
  for (int i = 0; i < 8; ++i) top[i] = NEGINF;

  tdm_load_to_lds(lds_off(kt[0]), kb, 2048, 0, 2048, 2048, 1);
  for (int pj = 0; pj < npair; ++pj) {
    if (pj + 1 < npair) {
      tdm_load_to_lds(lds_off(kt[(pj + 1) & 1]), kb + (size_t)(pj + 1) * 32 * DH,
                      2048, 0, 2048, 2048, 1);
      __builtin_amdgcn_s_wait_tensorcnt(1);
    } else {
      __builtin_amdgcn_s_wait_tensorcnt(0);
    }
    const __bf16* kc = kt[pj & 1];
    int j0 = pj * 32;
#pragma unroll
    for (int hh = 0; hh < 2; ++hh) {
      v16bf b0 = ld_row16(kc + (hh * 16 + l15) * DH + half * 16);
      v16bf b1 = ld_row16(kc + (hh * 16 + l15) * DH + half * 16 + 32);
      v8f c = {};
      c = wmma_bf16(qa0, b0, c);
      c = wmma_bf16(qa1, b1, c);
#pragma unroll
      for (int r = 0; r < 8; ++r) {
        int row = r + half * 8;
        int colg = j0 + hh * 16 + l15;
        float s = (colg <= m0 + row) ? c[r] * 0.125f : NEGINF;
        sm_s[row][hh * 16 + l15] = s;
      }
    }
    __syncthreads();
    int srow = l15;  // both halves duplicate rows 0..15 -> uniform flow
#pragma unroll
    for (int cc = 0; cc < 32; ++cc) {
      float v = sm_s[srow][cc];
      if (v > top[7]) {
        top[7] = v;
#pragma unroll
        for (int i = 7; i > 0; --i) {
          float x0 = top[i - 1], x1 = top[i];
          top[i - 1] = fmaxf(x0, x1);
          top[i] = fminf(x0, x1);
        }
      }
    }
    __syncthreads();
  }
  if (half == 0) { sm_thr[l15] = top[7]; sm_max[l15] = top[0]; }
  __syncthreads();
  float thr[8], mx[8];
#pragma unroll
  for (int r = 0; r < 8; ++r) {
    thr[r] = sm_thr[r + half * 8];
    mx[r] = sm_max[r + half * 8];
  }

  // ---------- Pass B: recompute, prune, exp, P*V ----------
  float dsum[8];
#pragma unroll
  for (int r = 0; r < 8; ++r) dsum[r] = 0.f;
  v8f oa[4] = {};

  tdm_load_to_lds(lds_off(kt[0]), kb, 2048, 0, 2048, 2048, 1);
  tdm_load_to_lds(lds_off(vt[0]), vb, 32, 64, (unsigned long long)T, 2048, 64);
  for (int pj = 0; pj < npair; ++pj) {
    if (pj + 1 < npair) {
      int jn = (pj + 1) * 32;
      tdm_load_to_lds(lds_off(kt[(pj + 1) & 1]), kb + (size_t)jn * DH,
                      2048, 0, 2048, 2048, 1);
      tdm_load_to_lds(lds_off(vt[(pj + 1) & 1]), vb + jn,
                      32, 64, (unsigned long long)T, 2048, 64);
      __builtin_amdgcn_s_wait_tensorcnt(2);
    } else {
      __builtin_amdgcn_s_wait_tensorcnt(0);
    }
    const __bf16* kc = kt[pj & 1];
    const __bf16* vc = vt[pj & 1];
    int j0 = pj * 32;
#pragma unroll
    for (int hh = 0; hh < 2; ++hh) {
      v16bf b0 = ld_row16(kc + (hh * 16 + l15) * DH + half * 16);
      v16bf b1 = ld_row16(kc + (hh * 16 + l15) * DH + half * 16 + 32);
      v8f c = {};
      c = wmma_bf16(qa0, b0, c);
      c = wmma_bf16(qa1, b1, c);
#pragma unroll
      for (int r = 0; r < 8; ++r) {
        int row = r + half * 8;
        int colg = j0 + hh * 16 + l15;
        float s = c[r] * 0.125f;
        bool keep = (colg <= m0 + row) && (s >= thr[r]);
        float p = keep ? __expf(s - mx[r]) : 0.f;
        dsum[r] += p;
        sp[row][hh * 16 + l15] = (__bf16)p;
      }
    }
    __syncthreads();
    v16bf pa = ld_afrag(&sp[l15][half * 16]);  // C-layout -> A-layout via LDS
    __syncthreads();
#pragma unroll
    for (int t4 = 0; t4 < 4; ++t4) {
      v16bf vf = ld_row16(vc + (t4 * 16 + l15) * 32 + half * 16);
      oa[t4] = wmma_bf16(pa, vf, oa[t4]);
    }
  }

#pragma unroll
  for (int r = 0; r < 8; ++r) {
    float s = dsum[r];
    s += __shfl_xor(s, 1);
    s += __shfl_xor(s, 2);
    s += __shfl_xor(s, 4);
    s += __shfl_xor(s, 8);
    dsum[r] = 1.0f / s;
  }
#pragma unroll
  for (int t4 = 0; t4 < 4; ++t4) {
#pragma unroll
    for (int r = 0; r < 8; ++r) {
      int row = m0 + r + half * 8;
      int col = h * DH + t4 * 16 + l15;
      ao[((size_t)b * T + row) * D + col] = (__bf16)(oa[t4][r] * dsum[r]);
    }
  }
}

// out[m][n] = sum_k A[m][k]*W[n][k]; 32x64 tile/wave, double-buffered; f32 out.
__global__ __launch_bounds__(256) void gemm_out_kernel(
    const __bf16* __restrict__ A, const __bf16* __restrict__ W,
    float* __restrict__ out) {
  const int K = 1024, NT = 16, N = 1024;
  int w = blockIdx.x * 8 + (threadIdx.x >> 5);
  int mt = w / NT, nt = w % NT;
  int lane = threadIdx.x & 31;
  int half = lane >> 4, l15 = lane & 15;
  int m0 = mt * 32, n0 = nt * 64;

  const __bf16* a0p = A + (size_t)(m0 + l15) * K + half * 16;
  const __bf16* a1p = a0p + (size_t)16 * K;
  const __bf16* bp[4];
#pragma unroll
  for (int t4 = 0; t4 < 4; ++t4)
    bp[t4] = W + (size_t)(n0 + t4 * 16 + l15) * K + half * 16;

  v8f acc[8] = {};
  v16bf a0c = ld_afrag(a0p), a1c = ld_afrag(a1p);
  v16bf bc[4];
#pragma unroll
  for (int t4 = 0; t4 < 4; ++t4) bc[t4] = ld_row16(bp[t4]);

  for (int k0 = 0; k0 < K; k0 += 32) {
    int kn = (k0 + 32 < K) ? k0 + 32 : 0;
    v16bf a0n = ld_afrag(a0p + kn);
    v16bf a1n = ld_afrag(a1p + kn);
    v16bf bn[4];
#pragma unroll
    for (int t4 = 0; t4 < 4; ++t4) bn[t4] = ld_row16(bp[t4] + kn);
#pragma unroll
    for (int t4 = 0; t4 < 4; ++t4) {
      acc[t4] = wmma_bf16(a0c, bc[t4], acc[t4]);
      acc[4 + t4] = wmma_bf16(a1c, bc[t4], acc[4 + t4]);
    }
    a0c = a0n; a1c = a1n;
#pragma unroll
    for (int t4 = 0; t4 < 4; ++t4) bc[t4] = bn[t4];
  }

#pragma unroll
  for (int mi = 0; mi < 2; ++mi) {
#pragma unroll
    for (int t4 = 0; t4 < 4; ++t4) {
#pragma unroll
      for (int r = 0; r < 8; ++r) {
        int row = m0 + mi * 16 + r + half * 8;
        int col = n0 + t4 * 16 + l15;
        out[(size_t)row * N + col] = acc[mi * 4 + t4][r];
      }
    }
  }
}

extern "C" void kernel_launch(void* const* d_in, const int* in_sizes, int n_in,
                              void* d_out, int out_size, void* d_ws, size_t ws_size,
                              hipStream_t stream) {
  (void)in_sizes; (void)n_in; (void)out_size; (void)ws_size;
  const float* x    = (const float*)d_in[0];   // (2,2048,1024) f32
  const float* wqkv = (const float*)d_in[1];   // (3072,1024) f32
  const float* wout = (const float*)d_in[2];   // (1024,1024) f32

  const size_t NX = 4u * 1024 * 1024;
  const size_t NWQ = 3u * 1024 * 1024;
  const size_t NWO = 1u * 1024 * 1024;
  const size_t NQ = 4u * 1024 * 1024;

  __bf16* xb  = (__bf16*)d_ws;
  __bf16* wqb = xb + NX;
  __bf16* wob = wqb + NWQ;
  __bf16* qws = wob + NWO;
  __bf16* kws = qws + NQ;
  __bf16* vtw = kws + NQ;      // transposed V: (B,H,dh,T)
  __bf16* aow = vtw + NQ;      // attention output bf16 (B,T,D)

  cvt_f32_bf16_kernel<<<(int)((NX + 255) / 256), 256, 0, stream>>>(x, xb, (int)NX);
  cvt_f32_bf16_kernel<<<(int)((NWQ + 255) / 256), 256, 0, stream>>>(wqkv, wqb, (int)NWQ);
  cvt_f32_bf16_kernel<<<(int)((NWO + 255) / 256), 256, 0, stream>>>(wout, wob, (int)NWO);

  // (4096/32)*(3072/64) = 6144 wave-tiles, 8 waves/block
  gemm_qkv_kernel<<<768, 256, 0, stream>>>(xb, wqb, qws, kws, vtw);
  // B*H*(T/16) = 4096 query tiles, 1 wave each
  attn_topk_kernel<<<4096, 32, 0, stream>>>(qws, kws, vtw, aow);
  // (4096/32)*(1024/64) = 2048 wave-tiles
  gemm_out_kernel<<<256, 256, 0, stream>>>(aow, wob, (float*)d_out);
}
